// GConvSE3_18743237279820
// MI455X (gfx1250) — matrix operand order repaired and checked by reading, not compiled, not run on hardware
//
#include <hip/hip_runtime.h>

typedef __attribute__((ext_vector_type(16))) _Float16 v16h;
typedef __attribute__((ext_vector_type(8)))  float    v8f;

#define NN   20000
#define EE   320000
#define EPW  16      // edges per destination node (dst = repeat(arange(N),16))
#define LNEPS 1e-5f

// ---------------------------------------------------------------------------
// One-time weight packer: converts rp_w2 (4x32x32) and w3_* to f16 WMMA
// B-fragments in workspace.
//  B-fragment layout used here (16-bit B, KxN = 32x16, contiguous k, same order as A):
//    lane l holds column N = l&15; element j (0..15) holds K = (l>>4)*16 + j
//  Fragment storage: [frag][lane][16 halfs] -> one 32B vector load per lane.
//  bf2: 8 frags  (pair*2 + ntile),  B[k][n] = w2[pair][ntile*16+n][k]
//  bf3: 24 frags (p0:t0-3, p1:t0-3, p2:t0-3, p3:t0-11), B[k][n] = w3[ntile*16+n][k]
// ---------------------------------------------------------------------------
__global__ void pack_bfrags(const float* __restrict__ rp_w2,
                            const float* __restrict__ w3_00,
                            const float* __restrict__ w3_01,
                            const float* __restrict__ w3_10,
                            const float* __restrict__ w3_11,
                            _Float16* __restrict__ bf2,
                            _Float16* __restrict__ bf3) {
  int idx = blockIdx.x * blockDim.x + threadIdx.x;
  const int total2 = 8 * 32 * 16;     // 4096
  const int total3 = 24 * 32 * 16;    // 12288
  if (idx < total2) {
    int j = idx & 15, lane = (idx >> 4) & 31, fp = idx >> 9;
    int pair = fp >> 1, tile = fp & 1;
    int n = lane & 15;
    int k = ((lane >> 4) << 4) + j;
    bf2[idx] = (_Float16)rp_w2[pair * 1024 + (tile * 16 + n) * 32 + k];
  } else if (idx < total2 + total3) {
    int t = idx - total2;
    int j = t & 15, lane = (t >> 4) & 31, f = t >> 9;
    int pair = (f < 12) ? (f >> 2) : 3;
    int tile = (f < 12) ? (f & 3) : (f - 12);
    const float* w3 = (pair == 0) ? w3_00 : (pair == 1) ? w3_01
                    : (pair == 2) ? w3_10 : w3_11;
    int n = lane & 15;
    int k = ((lane >> 4) << 4) + j;
    bf3[t] = (_Float16)w3[(tile * 16 + n) * 32 + k];
  }
}

// ---------------------------------------------------------------------------
// Fully fused SE(3) conv: one block per destination node (its 16 edges form
// one 16-row WMMA tile). 128 threads = 4 waves.
// ---------------------------------------------------------------------------
__global__ __launch_bounds__(128)
void gconv_node_kernel(const float* __restrict__ h0,
                       const float* __restrict__ h1,
                       const float* __restrict__ r,
                       const float* __restrict__ b00g,
                       const float* __restrict__ b01g,
                       const float* __restrict__ b10g,
                       const float* __restrict__ b11g,
                       const float* __restrict__ rp_w1,
                       const float* __restrict__ rp_b1,
                       const float* __restrict__ rp_g1,
                       const float* __restrict__ rp_be1,
                       const float* __restrict__ rp_b2,
                       const float* __restrict__ rp_g2,
                       const float* __restrict__ rp_be2,
                       const float* __restrict__ b3_00,
                       const float* __restrict__ b3_01,
                       const float* __restrict__ b3_10,
                       const float* __restrict__ b3_11,
                       const float* __restrict__ wself0,
                       const float* __restrict__ wself1,
                       const int*   __restrict__ src,
                       const _Float16* __restrict__ bf2,
                       const _Float16* __restrict__ bf3,
                       float* __restrict__ out) {
  __shared__ _Float16 X1[4][16][32];          // post layer1 (LN+ReLU), f16
  __shared__ float    X2r[4][16][32];         // raw layer2 output + bias
  __shared__ _Float16 X2h[4][16][32];         // post layer2 (LN+ReLU), f16
  __shared__ float    Rall[3 * 16 * 64 + 16 * 192];  // radial outputs R00..R11
  __shared__ float    S0[16][8];              // gathered h0[src]
  __shared__ float    S1[16][24];             // gathered h1[src]
  __shared__ float    Bs00[16];
  __shared__ float    Bs01[16][3];
  __shared__ float    Bs10[16][3];
  __shared__ float    Bs11[16][27];
  __shared__ float    Msg0[16][8];
  __shared__ float    Msg1[16][24];

  const int node = blockIdx.x;
  const int tid  = threadIdx.x;
  const int lane = tid & 31;
  const int wv   = tid >> 5;
  const int e0   = node * EPW;

  // ---- stage 0a: layer1 (x = r*w1+b1), LN, ReLU.  64 rows: (pair, edge) ----
  if (tid < 64) {
    int pair = tid >> 4, le = tid & 15;
    float rv = r[e0 + le];
    float x[32];
    float mu = 0.f;
#pragma unroll
    for (int h = 0; h < 32; ++h) {
      x[h] = rv * rp_w1[pair * 32 + h] + rp_b1[pair * 32 + h];
      mu += x[h];
    }
    mu *= (1.f / 32.f);
    float var = 0.f;
#pragma unroll
    for (int h = 0; h < 32; ++h) { float d = x[h] - mu; var += d * d; }
    var *= (1.f / 32.f);
    float is = rsqrtf(var + LNEPS);
#pragma unroll
    for (int h = 0; h < 32; ++h) {
      float y = (x[h] - mu) * is * rp_g1[pair * 32 + h] + rp_be1[pair * 32 + h];
      X1[pair][le][h] = (_Float16)fmaxf(y, 0.f);
    }
  }
  // ---- stage 0b: gather edge features & basis into LDS (all 128 threads) ----
  {
    int le = tid >> 3, q = tid & 7;   // 16 edges x 8 slots
    int e = e0 + le;
    int s = src[e];
    S0[le][q] = h0[s * 8 + q];
#pragma unroll
    for (int c = 0; c < 3; ++c) S1[le][q * 3 + c] = h1[s * 24 + q * 3 + c];
#pragma unroll
    for (int c = 0; c < 3; ++c) Bs11[le][q * 3 + c] = b11g[e * 27 + q * 3 + c];
    if (q == 0) {
      Bs11[le][24] = b11g[e * 27 + 24];
      Bs11[le][25] = b11g[e * 27 + 25];
      Bs11[le][26] = b11g[e * 27 + 26];
    }
    if (q == 1) { Bs01[le][0] = b01g[e*3]; Bs01[le][1] = b01g[e*3+1]; Bs01[le][2] = b01g[e*3+2]; }
    if (q == 2) { Bs10[le][0] = b10g[e*3]; Bs10[le][1] = b10g[e*3+1]; Bs10[le][2] = b10g[e*3+2]; }
    if (q == 3) { Bs00[le] = b00g[e]; }
  }
  __syncthreads();

  const int n16 = lane & 15;              // N (and A-row M) index
  const int mh  = (lane >> 4) << 3;       // C/D row-half offset
  const int kh  = (lane >> 4) << 3;       // A K-half offset

  // ---- stage 1: layer2 GEMM  X2 = X1 @ W2^T  (wave wv -> pair wv) ----
  {
    const _Float16* row = &X1[wv][n16][0];
    v16h a;
#pragma unroll
    for (int j = 0; j < 8; ++j) { a[j] = row[2 * kh + j]; a[j + 8] = row[2 * kh + 8 + j]; }
    v16h b0 = *(const v16h*)(bf2 + ((wv * 2 + 0) * 32 + lane) * 16);
    v16h b1 = *(const v16h*)(bf2 + ((wv * 2 + 1) * 32 + lane) * 16);
    v8f c0 = {}; v8f c1 = {};
    c0 = __builtin_amdgcn_wmma_f32_16x16x32_f16(false, a, false, b0, (short)0, c0, false, false);
    c1 = __builtin_amdgcn_wmma_f32_16x16x32_f16(false, a, false, b1, (short)0, c1, false, false);
#pragma unroll
    for (int v = 0; v < 8; ++v) {
      X2r[wv][mh + v][n16]      = c0[v] + rp_b2[wv * 32 + n16];
      X2r[wv][mh + v][16 + n16] = c1[v] + rp_b2[wv * 32 + 16 + n16];
    }
  }
  __syncthreads();

  // ---- stage 1b: LN2 + ReLU -> f16 ----
  if (tid < 64) {
    int pair = tid >> 4, le = tid & 15;
    float mu = 0.f;
#pragma unroll
    for (int h = 0; h < 32; ++h) mu += X2r[pair][le][h];
    mu *= (1.f / 32.f);
    float var = 0.f;
#pragma unroll
    for (int h = 0; h < 32; ++h) { float d = X2r[pair][le][h] - mu; var += d * d; }
    var *= (1.f / 32.f);
    float is = rsqrtf(var + LNEPS);
#pragma unroll
    for (int h = 0; h < 32; ++h) {
      float y = (X2r[pair][le][h] - mu) * is * rp_g2[pair * 32 + h] + rp_be2[pair * 32 + h];
      X2h[pair][le][h] = (_Float16)fmaxf(y, 0.f);
    }
  }
  __syncthreads();

  // ---- stage 2: layer3 GEMMs. 24 output N-tiles, 6 per wave ----
#pragma unroll
  for (int kk = 0; kk < 6; ++kk) {
    int f = wv * 6 + kk;
    int pair = (f < 12) ? (f >> 2) : 3;
    int tile = (f < 12) ? (f & 3) : (f - 12);
    const _Float16* row = &X2h[pair][n16][0];
    v16h a;
#pragma unroll
    for (int j = 0; j < 8; ++j) { a[j] = row[2 * kh + j]; a[j + 8] = row[2 * kh + 8 + j]; }
    v16h b = *(const v16h*)(bf3 + (f * 32 + lane) * 16);
    v8f d = {};
    d = __builtin_amdgcn_wmma_f32_16x16x32_f16(false, a, false, b, (short)0, d, false, false);
    const float* b3p = (pair == 0) ? b3_00 : (pair == 1) ? b3_01
                     : (pair == 2) ? b3_10 : b3_11;
    int stride = (pair < 3) ? 64 : 192;
    float* Rp = Rall + ((pair < 3) ? pair * 16 * 64 : 3 * 16 * 64);
    int col = tile * 16 + n16;
#pragma unroll
    for (int v = 0; v < 8; ++v) Rp[(mh + v) * stride + col] = d[v] + b3p[col];
  }
  __syncthreads();

  // ---- stage 3: per-edge tensor contractions + messages (thread = (edge, mo)) --
  {
    int le = tid >> 3, mo = tid & 7;
    const float* R00 = Rall + 0           + le * 64;
    const float* R01 = Rall + 16 * 64     + le * 64;
    const float* R10 = Rall + 2 * 16 * 64 + le * 64;
    const float* R11 = Rall + 3 * 16 * 64 + le * 192;
    const float* s0 = S0[le];
    const float* s1 = S1[le];
    // msg0[mo] = b00 * sum_mi R00[mo,mi]*s0[mi] + sum_mi R10[mo,mi]*dot(b10, s1[mi,:])
    float a00 = 0.f, a10 = 0.f;
#pragma unroll
    for (int mi = 0; mi < 8; ++mi) {
      a00 += R00[mo * 8 + mi] * s0[mi];
      float t1 = Bs10[le][0] * s1[mi * 3] + Bs10[le][1] * s1[mi * 3 + 1] +
                 Bs10[le][2] * s1[mi * 3 + 2];
      a10 += R10[mo * 8 + mi] * t1;
    }
    Msg0[le][mo] = Bs00[le] * a00 + a10;
    // msg1[mo*3+a] = b01[a]*u[mo] + sum_{f,b} P[mo,f,b]*b11[a,b,f]
    float u = 0.f;
    float P[3][3] = {};
#pragma unroll
    for (int mi = 0; mi < 8; ++mi) {
      u += R01[mo * 8 + mi] * s0[mi];
#pragma unroll
      for (int ff = 0; ff < 3; ++ff) {
        float rv = R11[(mo * 8 + mi) * 3 + ff];
#pragma unroll
        for (int bb = 0; bb < 3; ++bb) P[ff][bb] += rv * s1[mi * 3 + bb];
      }
    }
#pragma unroll
    for (int aa = 0; aa < 3; ++aa) {
      float m1 = Bs01[le][aa] * u;
#pragma unroll
      for (int ff = 0; ff < 3; ++ff)
#pragma unroll
        for (int bb = 0; bb < 3; ++bb)
          m1 += P[ff][bb] * Bs11[le][aa * 9 + bb * 3 + ff];
      Msg1[le][mo * 3 + aa] = m1;
    }
  }
  __syncthreads();

  // ---- stage 4: mean over the node's 16 edges + self-interaction ----
  if (tid < 32) {
    const float inv = 1.f / 16.f;   // in-degree is exactly 16 by construction
    if (tid < 8) {
      int mo = tid;
      float acc = 0.f;
#pragma unroll
      for (int le = 0; le < 16; ++le) acc += Msg0[le][mo];
      acc *= inv;
      float self = 0.f;
#pragma unroll
      for (int mi = 0; mi < 8; ++mi) self += wself0[mo * 8 + mi] * h0[node * 8 + mi];
      out[node * 8 + mo] = acc + self;
    } else {
      int idx = tid - 8;            // 0..23
      int mo = idx / 3, c = idx % 3;
      float acc = 0.f;
#pragma unroll
      for (int le = 0; le < 16; ++le) acc += Msg1[le][idx];
      acc *= inv;
      float self = 0.f;
#pragma unroll
      for (int mi = 0; mi < 8; ++mi) self += wself1[mo * 8 + mi] * h1[node * 24 + mi * 3 + c];
      out[NN * 8 + node * 24 + idx] = acc + self;
    }
  }
}

extern "C" void kernel_launch(void* const* d_in, const int* in_sizes, int n_in,
                              void* d_out, int out_size, void* d_ws, size_t ws_size,
                              hipStream_t stream) {
  (void)in_sizes; (void)n_in; (void)out_size; (void)ws_size;
  const float* h0     = (const float*)d_in[0];
  const float* h1     = (const float*)d_in[1];
  const float* r      = (const float*)d_in[2];
  const float* b00    = (const float*)d_in[3];
  const float* b01    = (const float*)d_in[4];
  const float* b10    = (const float*)d_in[5];
  const float* b11    = (const float*)d_in[6];
  const float* rp_w1  = (const float*)d_in[7];
  const float* rp_b1  = (const float*)d_in[8];
  const float* rp_g1  = (const float*)d_in[9];
  const float* rp_be1 = (const float*)d_in[10];
  const float* rp_w2  = (const float*)d_in[11];
  const float* rp_b2  = (const float*)d_in[12];
  const float* rp_g2  = (const float*)d_in[13];
  const float* rp_be2 = (const float*)d_in[14];
  const float* w3_00  = (const float*)d_in[15];
  const float* b3_00  = (const float*)d_in[16];
  const float* w3_01  = (const float*)d_in[17];
  const float* b3_01  = (const float*)d_in[18];
  const float* w3_10  = (const float*)d_in[19];
  const float* b3_10  = (const float*)d_in[20];
  const float* w3_11  = (const float*)d_in[21];
  const float* b3_11  = (const float*)d_in[22];
  const float* wself0 = (const float*)d_in[23];
  const float* wself1 = (const float*)d_in[24];
  const int*   src    = (const int*)d_in[25];
  // d_in[26] = dst: structure (repeat(arange(N),16)) exploited directly.

  _Float16* bf2 = (_Float16*)d_ws;                 // 8  * 32 * 16 halfs
  _Float16* bf3 = bf2 + 8 * 32 * 16;               // 24 * 32 * 16 halfs

  pack_bfrags<<<64, 256, 0, stream>>>(rp_w2, w3_00, w3_01, w3_10, w3_11, bf2, bf3);

  gconv_node_kernel<<<NN, 128, 0, stream>>>(
      h0, h1, r, b00, b01, b10, b11,
      rp_w1, rp_b1, rp_g1, rp_be1,
      rp_b2, rp_g2, rp_be2,
      b3_00, b3_01, b3_10, b3_11,
      wself0, wself1, src, bf2, bf3,
      (float*)d_out);
}
